// Multi_header_attention_12970801233980
// MI455X (gfx1250) — matrix operand order repaired and checked by reading, not compiled, not run on hardware
//
#include <hip/hip_runtime.h>
#include <hip/hip_bf16.h>

// ---------------------------------------------------------------------------
// MI455X (gfx1250) multi-head attention, bf16 WMMA, wave32,
// async global->LDS K/V staging (double buffered).
// ---------------------------------------------------------------------------

#define F_DIM  256
#define NSEQ   2048
#define BATCH  8
#define NH     4
#define HD     64

typedef __bf16 bf16;
typedef __attribute__((ext_vector_type(16))) __bf16 bf16x16;
typedef __attribute__((ext_vector_type(8)))  __bf16 bf16x8;
typedef __attribute__((ext_vector_type(4)))  __bf16 bf16x4;
typedef __attribute__((ext_vector_type(8)))  float  f32x8;
typedef __attribute__((ext_vector_type(4)))  float  f32x4;

__device__ __forceinline__ bf16 f2bf(float f) { return static_cast<bf16>(f); }

__device__ __forceinline__ bf16x16 join8(bf16x8 lo, bf16x8 hi) {
  return __builtin_shufflevector(lo, hi, 0,1,2,3,4,5,6,7,8,9,10,11,12,13,14,15);
}
__device__ __forceinline__ bf16x16 load16(const bf16* p) {   // 16B-aligned safe
  bf16x8 lo = *(const bf16x8*)p;
  bf16x8 hi = *(const bf16x8*)(p + 8);
  return join8(lo, hi);
}

// CDNA5 async global -> LDS copy, 16B per lane (tracked by ASYNCcnt).
__device__ __forceinline__ void async_ld16(unsigned lds_off, const bf16* gaddr) {
  asm volatile("global_load_async_to_lds_b128 %0, %1, off"
               :: "v"(lds_off), "v"(gaddr) : "memory");
}
__device__ __forceinline__ void wait_async0() {
  asm volatile("s_wait_asynccnt 0x0" ::: "memory");
}

// ---------------------------------------------------------------------------
// One-shot f32 -> bf16 weight conversion (W is 256x256 = 16384 float4s).
// ---------------------------------------------------------------------------
__global__ __launch_bounds__(256)
void cvt_w_kernel(const float* __restrict__ src, bf16* __restrict__ dst)
{
  const int i = blockIdx.x * 256 + threadIdx.x;
  f32x4 v = ((const f32x4*)src)[i];
  ((bf16x4*)dst)[i] = __builtin_convertvector(v, bf16x4);
}

// ---------------------------------------------------------------------------
// Shared LDS staging: X[b, f, n0:n0+16] -> xt[n-local][f] as bf16.
// ---------------------------------------------------------------------------
__device__ __forceinline__ void stage_x(const float* __restrict__ X, int b, int n0,
                                        bf16 (*xt)[264])
{
  const int f = threadIdx.x;
  const float4* src = (const float4*)(X + ((size_t)b * F_DIM + f) * NSEQ + n0);
#pragma unroll
  for (int c4 = 0; c4 < 4; ++c4) {
    float4 v = src[c4];
    xt[c4 * 4 + 0][f] = f2bf(v.x);
    xt[c4 * 4 + 1][f] = f2bf(v.y);
    xt[c4 * 4 + 2][f] = f2bf(v.z);
    xt[c4 * 4 + 3][f] = f2bf(v.w);
  }
  __syncthreads();
}

// ---------------------------------------------------------------------------
// Q/K projection: out[e,n] = sum_f W[e,f]*X[b,f,n] + bias[e], e = 4*d + h.
// Tile rows are the permuted set {e = 4*(d0+r)+h}, so each lane's accumulator
// holds 8 consecutive d -> single bf16x8 store into [bh][n][d].
// ---------------------------------------------------------------------------
__global__ __launch_bounds__(256)
void proj_qk_kernel(const float* __restrict__ X, const bf16* __restrict__ Wb,
                    const float* __restrict__ bias, bf16* __restrict__ out)
{
  __shared__ __align__(16) bf16 xt[16][264];
  const int blk = blockIdx.x;
  const int b   = blk >> 7;
  const int n0  = (blk & 127) << 4;
  stage_x(X, b, n0, xt);

  const int tid = threadIdx.x;
  const int wv = tid >> 5, lane = tid & 31;
  const int row = lane & 15, hi = lane >> 4;

#pragma unroll
  for (int i = 0; i < 2; ++i) {
    const int t  = (wv << 1) + i;          // 0..15
    const int h  = t >> 2;
    const int d0 = (t & 3) << 4;

    f32x8 acc;
#pragma unroll
    for (int r = 0; r < 8; ++r) acc[r] = bias[4 * (d0 + r + hi * 8) + h];

#pragma unroll
    for (int kk = 0; kk < 8; ++kk) {
      const int f0 = kk * 32;
      const bf16* wr = Wb + (size_t)(4 * (d0 + row) + h) * F_DIM + f0 + hi * 8;
      bf16x16 a = join8(*(const bf16x8*)wr, *(const bf16x8*)(wr + 16));
      bf16x16 bfr = join8(*(const bf16x8*)&xt[row][f0 + hi * 8], *(const bf16x8*)&xt[row][f0 + hi * 8 + 16]);
      acc = __builtin_amdgcn_wmma_f32_16x16x32_bf16(false, a, false, bfr,
                                                    (short)0, acc, false, false);
    }

    const int n = n0 + row;
    bf16x8 ov = __builtin_convertvector(acc, bf16x8);
    bf16* dst = out + (((size_t)(b * NH + h)) * NSEQ + n) * HD + d0 + hi * 8;
    *(bf16x8*)dst = ov;
  }
}

// ---------------------------------------------------------------------------
// Transposed projection: out^T[n][e] = sum_f X[f,n]*W[e,f] + bias[e].
// MODE 2: V layout bf16 [bh][d][m] (m contiguous per lane).
// MODE 3: final f32 [b][e][n]      (n contiguous per lane).
// ---------------------------------------------------------------------------
template<int MODE>
__global__ __launch_bounds__(256)
void proj_t_kernel(const float* __restrict__ X, const bf16* __restrict__ Wb,
                   const float* __restrict__ bias, void* __restrict__ out)
{
  __shared__ __align__(16) bf16 xt[16][264];
  const int blk = blockIdx.x;
  const int b   = blk >> 7;
  const int n0  = (blk & 127) << 4;
  stage_x(X, b, n0, xt);

  const int tid = threadIdx.x;
  const int wv = tid >> 5, lane = tid & 31;
  const int row = lane & 15, hi = lane >> 4;

#pragma unroll
  for (int i = 0; i < 2; ++i) {
    const int e0 = (((wv << 1) + i) << 4);
    const int e  = e0 + row;               // this lane's output column

    f32x8 acc;
    {
      const float bv = bias[e];
#pragma unroll
      for (int r = 0; r < 8; ++r) acc[r] = bv;
    }

#pragma unroll
    for (int kk = 0; kk < 8; ++kk) {
      const int f0 = kk * 32;
      bf16x16 a = join8(*(const bf16x8*)&xt[row][f0 + hi * 8],
                        *(const bf16x8*)&xt[row][f0 + hi * 8 + 16]);
      bf16x16 bfr = join8(*(const bf16x8*)(Wb + (size_t)e * F_DIM + f0 + hi * 8), *(const bf16x8*)(Wb + (size_t)e * F_DIM + f0 + hi * 8 + 16));
      acc = __builtin_amdgcn_wmma_f32_16x16x32_bf16(false, a, false, bfr,
                                                    (short)0, acc, false, false);
    }

    const int nbase = n0 + hi * 8;
    if (MODE == 3) {
      float* dst = (float*)out + ((size_t)b * F_DIM + e) * NSEQ + nbase;
      float4 lo4 = {acc[0], acc[1], acc[2], acc[3]};
      float4 hi4 = {acc[4], acc[5], acc[6], acc[7]};
      ((float4*)dst)[0] = lo4;
      ((float4*)dst)[1] = hi4;
    } else {
      const int d = e >> 2, h = e & 3;     // e = d*NH + h
      bf16* dst = (bf16*)out + (((size_t)(b * NH + h)) * HD + d) * NSEQ + nbase;
      *(bf16x8*)dst = __builtin_convertvector(acc, bf16x8);
    }
  }
}

// ---------------------------------------------------------------------------
// Flash attention. Block = 8 waves, all sharing one (b,h); wave w handles a
// 64-row query block. K/V tiles (32 m-columns) are staged global->LDS with
// async b128 copies, double-buffered, and shared by all 8 waves.
// grid = 128 blocks * 8 waves = 1024 waves = 32 bh * 32 n-blocks.
// ---------------------------------------------------------------------------
__global__ __launch_bounds__(256)
void attn_kernel(const bf16* __restrict__ q_ws, const bf16* __restrict__ k_ws,
                 const bf16* __restrict__ v_ws, float* __restrict__ x_ws)
{
  __shared__ __align__(16) bf16 kbuf[2][32][72];   // [buf][m][d], padded
  __shared__ __align__(16) bf16 vbuf[2][64][40];   // [buf][d][m], padded
  __shared__ __align__(16) bf16 pst[8][16][40];    // per-wave P staging

  const int tid = threadIdx.x;
  const int wv = tid >> 5, lane = tid & 31;
  const int gw = blockIdx.x * 8 + wv;
  const int bh = gw >> 5;            // same for all 8 waves in the block
  const int nb = gw & 31;
  const int b = bh >> 2, h = bh & 3;
  const int n0 = nb * 64;
  const int row = lane & 15, hi = lane >> 4;

  const bf16* qh = q_ws + (size_t)bh * NSEQ * HD;   // [n][d]
  const bf16* kh = k_ws + (size_t)bh * NSEQ * HD;   // [m][d]
  const bf16* vh = v_ws + (size_t)bh * HD * NSEQ;   // [d][m]

  // per-thread async staging assignments (16B chunks)
  const int km = tid >> 3, kc = tid & 7;   // K: 32 rows x 8 chunks
  const int vd = tid >> 2, vc = tid & 3;   // V: 64 rows x 4 chunks

  // Q A-fragments: 4 n-tiles x 2 d-halves (K=32 each)
  bf16x16 qf[4][2];
#pragma unroll
  for (int t = 0; t < 4; ++t) {
    const bf16* qr = qh + (size_t)(n0 + t * 16 + row) * HD;
#pragma unroll
    for (int dh = 0; dh < 2; ++dh) {
      const int dbase = dh * 32 + hi * 8;
      qf[t][dh] = join8(*(const bf16x8*)(qr + dbase),
                        *(const bf16x8*)(qr + dbase + 16));
    }
  }

  f32x8 acc[4][4];
  f32x8 mrow[4], lrow[4];
#pragma unroll
  for (int t = 0; t < 4; ++t) {
#pragma unroll
    for (int r = 0; r < 8; ++r) { mrow[t][r] = -3.0e38f; lrow[t][r] = 0.0f; }
#pragma unroll
    for (int dt = 0; dt < 4; ++dt)
#pragma unroll
      for (int r = 0; r < 8; ++r) acc[t][dt][r] = 0.0f;
  }

  // prologue: stage tile 0 into buffer 0
  async_ld16((unsigned)(uintptr_t)&kbuf[0][km][kc * 8], kh + (size_t)km * HD + kc * 8);
  async_ld16((unsigned)(uintptr_t)&vbuf[0][vd][vc * 8], vh + (size_t)vd * NSEQ + vc * 8);

  for (int it = 0; it < NSEQ / 32; ++it) {
    const int p = it & 1;
    wait_async0();        // this wave's copies into buf[p] have landed
    __syncthreads();      // everyone's copies have landed; buf[p^1] free

    if (it + 1 < NSEQ / 32) {   // stage next tile into the other buffer
      const int m1 = (it + 1) * 32;
      async_ld16((unsigned)(uintptr_t)&kbuf[p ^ 1][km][kc * 8],
                 kh + (size_t)(m1 + km) * HD + kc * 8);
      async_ld16((unsigned)(uintptr_t)&vbuf[p ^ 1][vd][vc * 8],
                 vh + (size_t)vd * NSEQ + m1 + vc * 8);
    }

    // K B-fragments from LDS: lane = col m, K(=d) contiguous
    bf16x16 kf[2][2];
#pragma unroll
    for (int mh = 0; mh < 2; ++mh)
#pragma unroll
      for (int dh = 0; dh < 2; ++dh)
        kf[mh][dh] = join8(*(const bf16x8*)&kbuf[p][mh * 16 + row][dh * 32 + hi * 8], *(const bf16x8*)&kbuf[p][mh * 16 + row][dh * 32 + hi * 8 + 16]);
    // V B-fragments from LDS: lane = col d, K(=m) contiguous
    bf16x16 vf[4];
#pragma unroll
    for (int dt = 0; dt < 4; ++dt)
      vf[dt] = join8(*(const bf16x8*)&vbuf[p][dt * 16 + row][hi * 8], *(const bf16x8*)&vbuf[p][dt * 16 + row][hi * 8 + 16]);

#pragma unroll
    for (int t = 0; t < 4; ++t) {
      f32x8 s0 = {0,0,0,0,0,0,0,0}, s1 = {0,0,0,0,0,0,0,0};
      s0 = __builtin_amdgcn_wmma_f32_16x16x32_bf16(false, qf[t][0], false, kf[0][0], (short)0, s0, false, false);
      s0 = __builtin_amdgcn_wmma_f32_16x16x32_bf16(false, qf[t][1], false, kf[0][1], (short)0, s0, false, false);
      s1 = __builtin_amdgcn_wmma_f32_16x16x32_bf16(false, qf[t][0], false, kf[1][0], (short)0, s1, false, false);
      s1 = __builtin_amdgcn_wmma_f32_16x16x32_bf16(false, qf[t][1], false, kf[1][1], (short)0, s1, false, false);

      // online softmax: rows live in 16-lane segments (C/D layout)
      f32x8 mnew, scale;
#pragma unroll
      for (int r = 0; r < 8; ++r) {
        float tm = fmaxf(s0[r], s1[r]);
        tm = fmaxf(tm, __shfl_xor(tm, 1, 16));
        tm = fmaxf(tm, __shfl_xor(tm, 2, 16));
        tm = fmaxf(tm, __shfl_xor(tm, 4, 16));
        tm = fmaxf(tm, __shfl_xor(tm, 8, 16));
        mnew[r]  = fmaxf(mrow[t][r], tm);
        scale[r] = __expf(mrow[t][r] - mnew[r]);
      }
#pragma unroll
      for (int r = 0; r < 8; ++r) {
        s0[r] = __expf(s0[r] - mnew[r]);
        s1[r] = __expf(s1[r] - mnew[r]);
        float rs = s0[r] + s1[r];
        rs += __shfl_xor(rs, 1, 16);
        rs += __shfl_xor(rs, 2, 16);
        rs += __shfl_xor(rs, 4, 16);
        rs += __shfl_xor(rs, 8, 16);
        lrow[t][r] = lrow[t][r] * scale[r] + rs;
      }
      mrow[t] = mnew;
#pragma unroll
      for (int dt = 0; dt < 4; ++dt)
#pragma unroll
        for (int r = 0; r < 8; ++r) acc[t][dt][r] *= scale[r];

      // D-layout -> A-layout via LDS (per-wave tile); packed hw conversion
      bf16x8 p0 = __builtin_convertvector(s0, bf16x8);
      bf16x8 p1 = __builtin_convertvector(s1, bf16x8);
#pragma unroll
      for (int r = 0; r < 8; ++r) {
        pst[wv][r + hi * 8][row]      = p0[r];
        pst[wv][r + hi * 8][row + 16] = p1[r];
      }
      const bf16* pr = &pst[wv][row][0];
      bf16x16 pf = join8(*(const bf16x8*)(pr + hi * 8),
                         *(const bf16x8*)(pr + hi * 8 + 16));
#pragma unroll
      for (int dt = 0; dt < 4; ++dt)
        acc[t][dt] = __builtin_amdgcn_wmma_f32_16x16x32_bf16(false, pf, false, vf[dt],
                                                             (short)0, acc[t][dt], false, false);
    }
  }

  // normalize (1/l) and post-softmax 1/sqrt(64)=1/8, write x_ws [b][f][n]
#pragma unroll
  for (int t = 0; t < 4; ++t) {
    f32x8 inv;
#pragma unroll
    for (int r = 0; r < 8; ++r) inv[r] = 1.0f / (lrow[t][r] * 8.0f);
#pragma unroll
    for (int dt = 0; dt < 4; ++dt) {
      f32x8 o;
#pragma unroll
      for (int r = 0; r < 8; ++r) o[r] = acc[t][dt][r] * inv[r];
      const int f = (dt * 16 + row) * NH + h;      // channel = d*NH + h
      float* dst = x_ws + ((size_t)b * F_DIM + f) * NSEQ + n0 + t * 16 + hi * 8;
      float4 lo4 = {o[0], o[1], o[2], o[3]};
      float4 hi4 = {o[4], o[5], o[6], o[7]};
      ((float4*)dst)[0] = lo4;
      ((float4*)dst)[1] = hi4;
    }
  }
}

// ---------------------------------------------------------------------------
extern "C" void kernel_launch(void* const* d_in, const int* in_sizes, int n_in,
                              void* d_out, int out_size, void* d_ws, size_t ws_size,
                              hipStream_t stream) {
  (void)in_sizes; (void)n_in; (void)out_size; (void)ws_size;
  const float* query = (const float*)d_in[0];
  const float* key_  = (const float*)d_in[1];
  const float* value = (const float*)d_in[2];
  const float* Wq = (const float*)d_in[3];
  const float* bq = (const float*)d_in[4];
  const float* Wk = (const float*)d_in[5];
  const float* bk = (const float*)d_in[6];
  const float* Wv = (const float*)d_in[7];
  const float* bv = (const float*)d_in[8];
  const float* Wm = (const float*)d_in[9];
  const float* bm = (const float*)d_in[10];

  char* ws = (char*)d_ws;
  bf16*  q_ws = (bf16*)(ws);                        //  8 MB  [bh][n][d]
  bf16*  k_ws = (bf16*)(ws + (size_t)( 8u << 20));  //  8 MB  [bh][m][d]
  bf16*  v_ws = (bf16*)(ws + (size_t)(16u << 20));  //  8 MB  [bh][d][m]
  float* x_ws = (float*)(ws + (size_t)(24u << 20)); // 16 MB  [b][f][n]
  bf16*  wq_b = (bf16*)(ws + (size_t)(40u << 20));            // 128 KB each
  bf16*  wk_b = (bf16*)(ws + (size_t)(40u << 20) + (128u << 10));
  bf16*  wv_b = (bf16*)(ws + (size_t)(40u << 20) + (256u << 10));
  bf16*  wm_b = (bf16*)(ws + (size_t)(40u << 20) + (384u << 10));

  dim3 cg(64), cb(256);
  cvt_w_kernel<<<cg, cb, 0, stream>>>(Wq, wq_b);
  cvt_w_kernel<<<cg, cb, 0, stream>>>(Wk, wk_b);
  cvt_w_kernel<<<cg, cb, 0, stream>>>(Wv, wv_b);
  cvt_w_kernel<<<cg, cb, 0, stream>>>(Wm, wm_b);

  dim3 pg(BATCH * (NSEQ / 16)), pb(256);
  proj_qk_kernel<<<pg, pb, 0, stream>>>(query, wq_b, bq, q_ws);
  proj_qk_kernel<<<pg, pb, 0, stream>>>(key_,  wk_b, bk, k_ws);
  proj_t_kernel<2><<<pg, pb, 0, stream>>>(value, wv_b, bv, (void*)v_ws);
  attn_kernel<<<dim3(128), dim3(256), 0, stream>>>(q_ws, k_ws, v_ws, x_ws);
  proj_t_kernel<3><<<pg, pb, 0, stream>>>(x_ws, wm_b, bm, d_out);
}
